// MINDModel_10814727651913
// MI455X (gfx1250) — matrix operand order repaired and checked by reading, not compiled, not run on hardware
//
#include <hip/hip_runtime.h>
#include <hip/hip_bf16.h>

// ---------------- problem constants ----------------
#define BATCH   16384
#define SEQL    100
#define DIM     32
#define NCAPS   4
#define H1      64
#define H2      32
#define KL      (NCAPS * SEQL)      // 400 routing logits

typedef __attribute__((ext_vector_type(16))) _Float16 v16h;
typedef __attribute__((ext_vector_type(8)))  float    v8f;

// ============================================================
// Kernel 1: mapped = gather(emb, behavior_items) @ S   (f16 out)
// One wave computes a 16-row tile of the [B*L, 32] x [32, 32] GEMM
// via two v_wmma_f32_16x16x32_f16 (N-tiles 0..15 and 16..31).
// ============================================================
__global__ void __launch_bounds__(256)
map_kernel(const int* __restrict__ items,
           const float* __restrict__ emb,
           const float* __restrict__ S,
           _Float16* __restrict__ mapped) {
    const int gwave = (blockIdx.x * blockDim.x + threadIdx.x) >> 5;   // tile id
    const int lane  = threadIdx.x & 31;
    const int half  = lane >> 4;          // 0 or 1
    const int m     = lane & 15;          // row within tile / column within N-tile
    const int ks    = half * 8;           // first 8-K group base for A
    const int kb    = half * 16;          // 16-K group base (unused)

    const long row  = (long)gwave * 16 + m;
    const int  item = items[row];
    const float* er = emb + (long)item * DIM;

    // A fragment: 16x32 f16 (ISA layout: lanes 0-15 K{0..7,16..23}, lanes 16-31 K{8..15,24..31})
    v16h a;
#pragma unroll
    for (int i = 0; i < 8; ++i) {
        a[i]     = (_Float16)er[ks + i];
        a[8 + i] = (_Float16)er[16 + ks + i];
    }
    // B fragments: S columns n0..n0+15 (lanes 0-15 K{0..7,16..23}, lanes 16-31 K{8..15,24..31})
    v16h b0, b1;
#pragma unroll
    for (int i = 0; i < 16; ++i) {
        b0[i] = (_Float16)S[(i < 8 ? ks + i : 8 + ks + i) * DIM + m];
        b1[i] = (_Float16)S[(i < 8 ? ks + i : 8 + ks + i) * DIM + 16 + m];
    }
    v8f z = {};
    v8f c0 = __builtin_amdgcn_wmma_f32_16x16x32_f16(false, a, false, b0, (short)0, z, false, false);
    v8f c1 = __builtin_amdgcn_wmma_f32_16x16x32_f16(false, a, false, b1, (short)0, z, false, false);

    // D layout: VGPR i -> row (half*8+i), col (lane&15) within N-tile
    _Float16* out = mapped + (long)gwave * 16 * DIM;
#pragma unroll
    for (int i = 0; i < 8; ++i) {
        const int rm = half * 8 + i;
        out[rm * DIM + m]      = (_Float16)c0[i];
        out[rm * DIM + 16 + m] = (_Float16)c1[i];
    }
}

// ============================================================
// Kernel 2: one routing iteration.
//   FINAL=false : compute caps, accumulate block-partial delta -> part[]
//   FINAL=true  : write caps (f16) for the MLP kernel, no delta
// One wave handles 8 batch rows sequentially; 8 waves / block.
// ============================================================
#define RWAVES 8
#define ROWS_PER_WAVE 8
#define RBLOCKS (BATCH / (RWAVES * ROWS_PER_WAVE))   // 256

template <bool FINAL>
__global__ void __launch_bounds__(256)
routing_kernel(const _Float16* __restrict__ mapped,
               const int* __restrict__ seq_len,
               const float* __restrict__ logits_in,
               float* __restrict__ part,          // [gridDim.x * 400] partials (!FINAL)
               _Float16* __restrict__ caps_out) { // [B*K*32] f16 (FINAL)
    __shared__ float sLog[KL];
    __shared__ float sDelta[KL];
    __shared__ float sW[RWAVES][KL];
    __shared__ float sCaps[RWAVES][NCAPS * DIM];

    const int tid  = threadIdx.x;
    const int w    = tid >> 5;
    const int lane = tid & 31;

    for (int i = tid; i < KL; i += 256) {
        sLog[i] = logits_in[i];
        if (!FINAL) sDelta[i] = 0.f;
    }
    __syncthreads();

    const int rowBase = blockIdx.x * (RWAVES * ROWS_PER_WAVE) + w * ROWS_PER_WAVE;

    for (int r = 0; r < ROWS_PER_WAVE; ++r) {
        const int b = rowBase + r;
        const int n = seq_len[b];
        const _Float16* mrow = mapped + (long)b * SEQL * DIM;

        // ---- masked softmax over l for each capsule k ----
#pragma unroll
        for (int k = 0; k < NCAPS; ++k) {
            float v[4];
            float mx = -3.0e38f;
#pragma unroll
            for (int j = 0; j < 4; ++j) {
                const int l = lane + 32 * j;
                float t = (l < SEQL && l < n) ? sLog[k * SEQL + l] : -3.0e38f;
                v[j] = t;
                mx = fmaxf(mx, t);
            }
            for (int off = 16; off; off >>= 1) mx = fmaxf(mx, __shfl_xor(mx, off, 32));
            float s = 0.f;
#pragma unroll
            for (int j = 0; j < 4; ++j) {
                float e = (v[j] <= -1.0e38f) ? 0.f : __expf(v[j] - mx);
                v[j] = e;
                s += e;
            }
            for (int off = 16; off; off >>= 1) s += __shfl_xor(s, off, 32);
            const float inv = 1.0f / s;
#pragma unroll
            for (int j = 0; j < 4; ++j) {
                const int l = lane + 32 * j;
                if (l < SEQL) sW[w][k * SEQL + l] = v[j] * inv;
            }
        }
        __syncthreads();

        // ---- Z[k,:] = sum_l w[k,l] * mapped[l,:]  (lane == dim) ----
        float Z[NCAPS] = {0.f, 0.f, 0.f, 0.f};
        for (int l = 0; l < SEQL; ++l) {
            const float mv = (float)mrow[l * DIM + lane];
            const float* wk = &sW[w][l];
            Z[0] += wk[0]   * mv;
            Z[1] += wk[100] * mv;
            Z[2] += wk[200] * mv;
            Z[3] += wk[300] * mv;
        }
        // ---- squash ----
#pragma unroll
        for (int k = 0; k < NCAPS; ++k) {
            float sq = Z[k] * Z[k];
            for (int off = 16; off; off >>= 1) sq += __shfl_xor(sq, off, 32);
            const float scale = (sq / (1.f + sq)) * rsqrtf(sq + 1e-8f);
            const float cap = scale * Z[k];
            if (FINAL) caps_out[((long)b * NCAPS + k) * DIM + lane] = (_Float16)cap;
            else       sCaps[w][k * DIM + lane] = cap;
        }

        if (!FINAL) {
            __syncthreads();
            // ---- delta[k,l] = caps[k,:] . mapped[l,:]  (lane-parallel over l) ----
#pragma unroll
            for (int j = 0; j < 4; ++j) {
                const int l = lane + 32 * j;
                if (l < SEQL) {
                    const _Float16* ml = mrow + l * DIM;
                    float d0 = 0.f, d1 = 0.f, d2 = 0.f, d3 = 0.f;
                    for (int d = 0; d < DIM; ++d) {
                        const float mv = (float)ml[d];
                        d0 += sCaps[w][d]      * mv;
                        d1 += sCaps[w][32 + d] * mv;
                        d2 += sCaps[w][64 + d] * mv;
                        d3 += sCaps[w][96 + d] * mv;
                    }
                    atomicAdd(&sDelta[l],       d0);
                    atomicAdd(&sDelta[100 + l], d1);
                    atomicAdd(&sDelta[200 + l], d2);
                    atomicAdd(&sDelta[300 + l], d3);
                }
            }
        }
        __syncthreads();
    }

    if (!FINAL) {
        for (int i = tid; i < KL; i += 256)
            part[(long)blockIdx.x * KL + i] = sDelta[i];
    }
}

// ============================================================
// Kernel 3: deterministic fixed-order reduction of block partials:
//   logits_out[i] = logits_in[i] + sum_blk part[blk*400 + i]
// ============================================================
__global__ void reduce_logits(const float* __restrict__ logits_in,
                              const float* __restrict__ part,
                              float* __restrict__ logits_out) {
    const int i = threadIdx.x;
    if (i < KL) {
        float s = logits_in[i];
        for (int blk = 0; blk < RBLOCKS; ++blk) s += part[blk * KL + i];
        logits_out[i] = s;
    }
}

// ============================================================
// Kernel 4: MLP (two WMMA GEMMs) + label-aware attention + argmax
// One wave handles 16 capsule rows (= 4 batch rows x 4 capsules).
// ============================================================
#define MWAVES 8

__global__ void __launch_bounds__(256)
mlp_kernel(const _Float16* __restrict__ caps,
           const float* __restrict__ W1, const float* __restrict__ b1,
           const float* __restrict__ W2, const float* __restrict__ b2,
           const int* __restrict__ target_items,
           const float* __restrict__ emb,
           float* __restrict__ out) {
    __shared__ _Float16 u1s[MWAVES][16 * H1];

    const int tid  = threadIdx.x;
    const int w    = tid >> 5;
    const int lane = tid & 31;
    const int tile = blockIdx.x * MWAVES + w;
    const int half = lane >> 4;
    const int m    = lane & 15;
    const int ks   = half * 8;
    const int kb   = half * 16;
    const int rowBase = tile * 16;

    // A1: caps tile [16 x 32] f16 (row-major in memory -> A layout)
    const _Float16* cr = caps + (long)(rowBase + m) * DIM;
    v16h a1;
#pragma unroll
    for (int i = 0; i < 8; ++i) {
        a1[i]     = cr[ks + i];
        a1[8 + i] = cr[16 + ks + i];
    }

    // ---- layer 1: [16x32] @ [32x64] -> 4 N-tiles ----
    v8f c[4];
#pragma unroll
    for (int t = 0; t < 4; ++t) {
        const int n0 = t * 16;
        v16h bf;
#pragma unroll
        for (int i = 0; i < 16; ++i) bf[i] = (_Float16)W1[(i < 8 ? ks + i : 8 + ks + i) * H1 + n0 + m];
        v8f z = {};
        c[t] = __builtin_amdgcn_wmma_f32_16x16x32_f16(false, a1, false, bf, (short)0, z, false, false);
    }
    // bias + relu -> LDS (f16, row-major 16x64) for A2 restage
#pragma unroll
    for (int t = 0; t < 4; ++t) {
        const int col = t * 16 + m;
        const float bb = b1[col];
#pragma unroll
        for (int i = 0; i < 8; ++i) {
            const int rm = half * 8 + i;
            float v = c[t][i] + bb;
            u1s[w][rm * H1 + col] = (_Float16)(v > 0.f ? v : 0.f);
        }
    }
    __syncthreads();

    // A2 fragments (K = 64 split into 2 x 32)
    v16h a2a, a2b;
#pragma unroll
    for (int i = 0; i < 8; ++i) {
        a2a[i]     = u1s[w][m * H1 + ks + i];
        a2a[8 + i] = u1s[w][m * H1 + 16 + ks + i];
        a2b[i]     = u1s[w][m * H1 + 32 + ks + i];
        a2b[8 + i] = u1s[w][m * H1 + 48 + ks + i];
    }

    // ---- layer 2: [16x64] @ [64x32] -> 2 N-tiles, chained K ----
    v8f d[2];
#pragma unroll
    for (int nt = 0; nt < 2; ++nt) {
        const int n0 = nt * 16;
        v16h bA, bB;
#pragma unroll
        for (int i = 0; i < 16; ++i) {
            bA[i] = (_Float16)W2[(i < 8 ? ks + i : 8 + ks + i) * H2 + n0 + m];
            bB[i] = (_Float16)W2[(32 + (i < 8 ? ks + i : 8 + ks + i)) * H2 + n0 + m];
        }
        v8f z = {};
        v8f t0 = __builtin_amdgcn_wmma_f32_16x16x32_f16(false, a2a, false, bA, (short)0, z,  false, false);
        d[nt]  = __builtin_amdgcn_wmma_f32_16x16x32_f16(false, a2b, false, bB, (short)0, t0, false, false);
    }

    // bias + relu
    float u2a[8], u2b[8];
    {
        const float bb0 = b2[m], bb1 = b2[16 + m];
#pragma unroll
        for (int i = 0; i < 8; ++i) {
            const float v0 = d[0][i] + bb0;
            const float v1 = d[1][i] + bb1;
            u2a[i] = v0 > 0.f ? v0 : 0.f;
            u2b[i] = v1 > 0.f ? v1 : 0.f;
        }
    }

    // ---- label-aware attention weight: row . tgt_emb[b] ----
    float wpart[8];
#pragma unroll
    for (int i = 0; i < 8; ++i) {
        const int rm   = half * 8 + i;
        const int bidx = (rowBase + rm) >> 2;
        const int it   = target_items[bidx];
        const float t0 = emb[(long)it * DIM + m];
        const float t1 = emb[(long)it * DIM + 16 + m];
        wpart[i] = u2a[i] * t0 + u2b[i] * t1;
    }
    // reduce across the 16 lanes that hold this row's columns
#pragma unroll
    for (int i = 0; i < 8; ++i)
        for (int off = 8; off; off >>= 1) wpart[i] += __shfl_xor(wpart[i], off, 32);

    // ---- first-max argmax over the 4 capsules of each batch row; emit winner ----
#pragma unroll
    for (int g = 0; g < 2; ++g) {
        const int base = g * 4;
        int kbst = 0;
        float best = wpart[base];
#pragma unroll
        for (int j = 1; j < 4; ++j)
            if (wpart[base + j] > best) { best = wpart[base + j]; kbst = j; }
        const int sel = base + kbst;
        float o0 = u2a[0], o1 = u2b[0];
#pragma unroll
        for (int j = 1; j < 8; ++j)
            if (sel == j) { o0 = u2a[j]; o1 = u2b[j]; }
        const int bidx = (rowBase + half * 8 + base) >> 2;
        out[(long)bidx * H2 + m]      = o0;
        out[(long)bidx * H2 + 16 + m] = o1;
    }
}

// ============================================================
// Launch
// ============================================================
extern "C" void kernel_launch(void* const* d_in, const int* in_sizes, int n_in,
                              void* d_out, int out_size, void* d_ws, size_t ws_size,
                              hipStream_t stream) {
    (void)in_sizes; (void)n_in; (void)out_size; (void)ws_size;

    const int*   behavior_items = (const int*)d_in[0];
    const int*   target_items   = (const int*)d_in[1];
    const int*   seq_len        = (const int*)d_in[2];
    const float* emb_table      = (const float*)d_in[3];
    const float* S              = (const float*)d_in[4];
    const float* routing_logits = (const float*)d_in[5];
    const float* W1             = (const float*)d_in[6];
    const float* b1             = (const float*)d_in[7];
    const float* W2             = (const float*)d_in[8];
    const float* b2             = (const float*)d_in[9];
    float* out = (float*)d_out;

    // workspace layout
    char* ws = (char*)d_ws;
    _Float16* mapped = (_Float16*)ws;                                   // B*L*32 f16 = 104,857,600 B
    size_t off = (size_t)BATCH * SEQL * DIM * sizeof(_Float16);
    _Float16* capsb = (_Float16*)(ws + off);                            // B*K*32 f16 = 4,194,304 B
    off += (size_t)BATCH * NCAPS * DIM * sizeof(_Float16);
    float* part = (float*)(ws + off);                                   // 256*400 f32
    off += (size_t)RBLOCKS * KL * sizeof(float);
    float* bufA = (float*)(ws + off);  off += 512 * sizeof(float);      // 400 (+pad)
    float* bufB = (float*)(ws + off);

    // 1) mapped = gather @ S        (WMMA)
    {
        const int tiles = BATCH * SEQL / 16;        // 102,400
        map_kernel<<<tiles / 8, 256, 0, stream>>>(behavior_items, emb_table, S, mapped);
    }
    // 2) routing iteration 0: logitsA = routing_logits + sum_b delta0
    routing_kernel<false><<<RBLOCKS, 256, 0, stream>>>(mapped, seq_len, routing_logits, part, nullptr);
    reduce_logits<<<1, 512, 0, stream>>>(routing_logits, part, bufA);
    // 3) routing iteration 1: logitsB = logitsA + sum_b delta1
    routing_kernel<false><<<RBLOCKS, 256, 0, stream>>>(mapped, seq_len, bufA, part, nullptr);
    reduce_logits<<<1, 512, 0, stream>>>(bufA, part, bufB);
    // 4) routing iteration 2 (final caps; delta is dead in the reference)
    routing_kernel<true><<<RBLOCKS, 256, 0, stream>>>(mapped, seq_len, bufB, nullptr, capsb);
    // 5) MLP + label-aware attention   (WMMA)
    mlp_kernel<<<(BATCH * NCAPS / 16) / MWAVES, 256, 0, stream>>>(
        capsb, W1, b1, W2, b2, target_items, emb_table, out);
}
